// LSG_NET_70686571757678
// MI455X (gfx1250) — matrix operand order repaired and checked by reading, not compiled, not run on hardware
//
#include <hip/hip_runtime.h>
#include <hip/hip_bf16.h>
#include <cstdint>

typedef __bf16 bf16_t;
typedef __attribute__((ext_vector_type(16))) __bf16 v16bf;
typedef __attribute__((ext_vector_type(8)))  __bf16 v8bf;
typedef __attribute__((ext_vector_type(8)))  float   v8f;

// Problem dims
#define BATCH 8
#define SEQ   32
#define CHAN  512
#define HW    49          // 7x7
#define DFLAT (CHAN*HW)   // 25088
#define MDIM  2048        // 4*HID
#define KDIM  13824       // 1536*9
#define NREAL 392         // 8*49
#define NPAD  400
// GEMM tiling: 256 threads = 8 waves; each wave owns 2 M-frags x 5 N-frags
#define MT    256
#define MHALF 128
#define NT    80
#define KT    32

// ---------------- workspace layout (bytes) ----------------
static const size_t OFF_A    = 0;                                   // bf16 weights [2048][13824]
static const size_t OFF_B    = OFF_A   + (size_t)MDIM * KDIM * 2;   // bf16 im2col  [400][13824]
static const size_t OFF_G    = OFF_B   + (size_t)NPAD * KDIM * 2;   // f32 gates    [2048][400]
static const size_t OFF_HBF  = OFF_G   + (size_t)MDIM * NPAD * 4;   // bf16 h_prev  [512][392]
static const size_t OFF_C    = OFF_HBF + (size_t)CHAN * NREAL * 2;  // f32 c state  [512][392]
static const size_t OFF_HS   = OFF_C   + (size_t)CHAN * NREAL * 4;  // f32 hs [8][31][512][49]
static const size_t OFF_DOT  = OFF_HS  + (size_t)BATCH * 31 * DFLAT * 4; // [8][32][31]
static const size_t OFF_NX   = OFF_DOT + (size_t)BATCH * 32 * 31 * 4;    // [256]
static const size_t OFF_NH   = OFF_NX  + 256 * 4;                        // [248] (pad 256)
static const size_t OFF_WSUM = OFF_NH  + 256 * 4;                        // [256]

// CDNA5 async copy global->LDS (tracked by ASYNCcnt)
#define ASYNC_LD_B128(ldsoff, gptr)                                     \
  asm volatile("global_load_async_to_lds_b128 %0, %1, off"              \
               :: "v"(ldsoff), "v"(gptr) : "memory")
#define WAIT_ASYNC0() asm volatile("s_wait_asynccnt 0" ::: "memory")

__device__ __forceinline__ uint32_t lds_addr32(const void* p) {
  // generic pointer to LDS: low 32 bits are the wave-relative LDS offset
  return (uint32_t)(uintptr_t)p;
}

// ---------------- fp32 -> bf16 weight convert ----------------
__global__ void cvt_bf16_kernel(const float* __restrict__ src,
                                bf16_t* __restrict__ dst, size_t nelem) {
  size_t i = ((size_t)blockIdx.x * blockDim.x + threadIdx.x) * 4;
  if (i >= nelem) return;
#pragma unroll
  for (int j = 0; j < 4; ++j) dst[i + j] = (bf16_t)src[i + j];
}

// ---------------- zero h/c state ----------------
__global__ void init_state_kernel(float* __restrict__ cst, bf16_t* __restrict__ hbf) {
  int i = blockIdx.x * blockDim.x + threadIdx.x;
  if (i < CHAN * NREAL) { cst[i] = 0.f; hbf[i] = (bf16_t)0.f; }
}

// ---------------- im2col: build B matrix [n][k], k contiguous ----------------
__global__ void im2col_kernel(const float* __restrict__ features,
                              const bf16_t* __restrict__ hprev,
                              bf16_t* __restrict__ Bm, int t) {
  size_t idx = (size_t)blockIdx.x * blockDim.x + threadIdx.x;
  if (idx >= (size_t)NREAL * KDIM) return;
  int n = (int)(idx / KDIM);
  int k = (int)(idx % KDIM);
  int b = n / HW, pix = n % HW, oh = pix / 7, ow = pix % 7;
  int ci = k / 9, r = k % 9, kh = r / 3, kw = r % 3;
  int ih = oh + kh - 1, iw = ow + kw - 1;
  float v = 0.f;
  if (ih >= 0 && ih < 7 && iw >= 0 && iw < 7) {
    int p = ih * 7 + iw;
    if (ci < CHAN)
      v = features[(((size_t)b * SEQ + t) * CHAN + ci) * HW + p];
    else if (ci < 2 * CHAN)
      v = features[(((size_t)b * SEQ + t + 1) * CHAN + (ci - CHAN)) * HW + p];
    else
      v = (float)hprev[(size_t)(ci - 2 * CHAN) * NREAL + b * HW + p];
  }
  Bm[(size_t)n * KDIM + k] = (bf16_t)v;
}

// ---------------- WMMA bf16 GEMM: gates[2048][400] = A[2048][K] * B[400][K]^T ----------------
__global__ __launch_bounds__(256) void gemm_gates_kernel(
    const bf16_t* __restrict__ A, const bf16_t* __restrict__ Bm,
    float* __restrict__ gates) {
  __shared__ __align__(64) bf16_t sA[2][MT * KT];   // row-major, K contiguous (64B rows)
  __shared__ __align__(64) bf16_t sB[2][NT * KT];   // per-N column, K contiguous

  const int tid = threadIdx.x;
  const int wave = tid >> 5, lane = tid & 31;
  const int m_base = blockIdx.x * MT;
  const int n_base = blockIdx.y * NT;

  v8f acc[2][5] = {};

  auto load_stage = [&](int buf, int k0) {
    // A tile: 256 rows x 64B = 1024 x 16B chunks, 4 per thread
#pragma unroll
    for (int it = 0; it < 4; ++it) {
      int chunk = tid + it * 256;
      int row = chunk >> 2, c16 = chunk & 3;
      const bf16_t* src = A + (size_t)(m_base + row) * KDIM + k0 + c16 * 8;
      ASYNC_LD_B128(lds_addr32(&sA[buf][row * KT + c16 * 8]), src);
    }
    // B tile: 80 rows x 64B = 320 x 16B chunks
    {
      int row = tid >> 2, c16 = tid & 3;
      const bf16_t* src = Bm + (size_t)(n_base + row) * KDIM + k0 + c16 * 8;
      ASYNC_LD_B128(lds_addr32(&sB[buf][row * KT + c16 * 8]), src);
    }
    if (tid < 64) {
      int chunk = 256 + tid;
      int row = chunk >> 2, c16 = chunk & 3;
      const bf16_t* src = Bm + (size_t)(n_base + row) * KDIM + k0 + c16 * 8;
      ASYNC_LD_B128(lds_addr32(&sB[buf][row * KT + c16 * 8]), src);
    }
    // L2 prefetch two stages ahead (global_prefetch_b8)
    if (k0 + 2 * KT < KDIM)
      __builtin_prefetch(A + (size_t)(m_base + tid) * KDIM + k0 + 2 * KT, 0, 1);
  };

  load_stage(0, 0);
  WAIT_ASYNC0();
  __syncthreads();

  const int nk = KDIM / KT;  // 432
  for (int kt = 0; kt < nk; ++kt) {
    const int buf = kt & 1;
    if (kt + 1 < nk) load_stage(buf ^ 1, (kt + 1) * KT);

    // Two A fragments 16x32 (lanes 0-15 hold K 0-15,
    // lanes 16-31 hold K 16-31, matching B)
    v16bf afrag[2];
#pragma unroll
    for (int mf = 0; mf < 2; ++mf) {
      const int arow = mf * MHALF + wave * 16 + (lane & 15);
      const int khalf = (lane >> 4) * 16;
      const bf16_t* ap = &sA[buf][arow * KT + khalf];
      v8bf alo = *(const v8bf*)ap;
      v8bf ahi = *(const v8bf*)(ap + 8);
      afrag[mf] = __builtin_shufflevector(alo, ahi, 0, 1, 2, 3, 4, 5, 6, 7,
                                          8, 9, 10, 11, 12, 13, 14, 15);
    }
#pragma unroll
    for (int nf = 0; nf < 5; ++nf) {
      // B fragment 32x16: lane holds 16 contiguous K for its N column;
      // reused by both M fragments (halves B-side LDS traffic per FLOP)
      const int bcol = nf * 16 + (lane & 15);
      const bf16_t* bp = &sB[buf][bcol * KT + (lane >> 4) * 16];
      v16bf bfrag = *(const v16bf*)bp;
      acc[0][nf] = __builtin_amdgcn_wmma_f32_16x16x32_bf16(
          false, afrag[0], false, bfrag, (short)0, acc[0][nf], false, false);
      acc[1][nf] = __builtin_amdgcn_wmma_f32_16x16x32_bf16(
          false, afrag[1], false, bfrag, (short)0, acc[1][nf], false, false);
    }

    WAIT_ASYNC0();
    __syncthreads();
  }

  // C/D layout: VGPR r -> M = r (+8 for lanes 16-31), N = lane%16
#pragma unroll
  for (int mf = 0; mf < 2; ++mf) {
    const int mrow = m_base + mf * MHALF + wave * 16 + ((lane >> 4) << 3);
#pragma unroll
    for (int nf = 0; nf < 5; ++nf) {
      const int col = n_base + nf * 16 + (lane & 15);
#pragma unroll
      for (int r = 0; r < 8; ++r)
        gates[(size_t)(mrow + r) * NPAD + col] = acc[mf][nf][r];
    }
  }
}

// ---------------- LSTM pointwise: gates -> c,h ----------------
__global__ void lstm_pointwise_kernel(const float* __restrict__ gates,
                                      const float* __restrict__ bias,
                                      float* __restrict__ cst,
                                      bf16_t* __restrict__ hbf,
                                      float* __restrict__ hs, int t) {
  int idx = blockIdx.x * blockDim.x + threadIdx.x;
  if (idx >= CHAN * NREAL) return;
  int c = idx / NREAL, n = idx % NREAL;
  int b = n / HW, pix = n % HW;
  float gi = gates[(size_t)(c         ) * NPAD + n] + bias[c];
  float gf = gates[(size_t)(c +   CHAN) * NPAD + n] + bias[c + CHAN];
  float go = gates[(size_t)(c + 2*CHAN) * NPAD + n] + bias[c + 2*CHAN];
  float gg = gates[(size_t)(c + 3*CHAN) * NPAD + n] + bias[c + 3*CHAN];
  float si = 1.f / (1.f + __expf(-gi));
  float sf = 1.f / (1.f + __expf(-gf));
  float so = 1.f / (1.f + __expf(-go));
  float cv = sf * cst[idx] + si * tanhf(gg);
  cst[idx] = cv;
  float hv = so * tanhf(cv);
  hbf[idx] = (bf16_t)hv;
  hs[(((size_t)b * 31 + t) * CHAN + c) * HW + pix] = hv;
}

// ---------------- cosine-sim dots + |f| norms ----------------
__global__ __launch_bounds__(256) void simdot_kernel(
    const float* __restrict__ features, const float* __restrict__ hs,
    float* __restrict__ dot, float* __restrict__ nx) {
  int s = blockIdx.x, b = blockIdx.y;
  int tid = threadIdx.x, wave = tid >> 5, lane = tid & 31;
  float acc[31];
#pragma unroll
  for (int t = 0; t < 31; ++t) acc[t] = 0.f;
  float ff = 0.f;
  const float* f  = features + ((size_t)b * SEQ + s) * DFLAT;
  const float* hb = hs + (size_t)b * 31 * DFLAT;
  for (int d = tid; d < DFLAT; d += 256) {
    float fv = f[d];
    ff += fv * fv;
#pragma unroll
    for (int t = 0; t < 31; ++t) acc[t] += fv * hb[(size_t)t * DFLAT + d];
  }
#pragma unroll
  for (int t = 0; t < 31; ++t)
    for (int o = 16; o; o >>= 1) acc[t] += __shfl_xor(acc[t], o, 32);
  for (int o = 16; o; o >>= 1) ff += __shfl_xor(ff, o, 32);
  __shared__ float red[8][32];
  if (lane == 0) {
#pragma unroll
    for (int t = 0; t < 31; ++t) red[wave][t] = acc[t];
    red[wave][31] = ff;
  }
  __syncthreads();
  if (tid < 32) {
    float v = 0.f;
#pragma unroll
    for (int w = 0; w < 8; ++w) v += red[w][tid];
    if (tid < 31) dot[((size_t)b * SEQ + s) * 31 + tid] = v;
    else          nx[b * SEQ + s] = sqrtf(v);
  }
}

// ---------------- |h| norms ----------------
__global__ void nh_kernel(const float* __restrict__ hs, float* __restrict__ nh) {
  int row = blockIdx.x;  // b*31 + t
  int tid = threadIdx.x;
  const float* h = hs + (size_t)row * DFLAT;
  float sm = 0.f;
  for (int d = tid; d < DFLAT; d += 256) { float v = h[d]; sm += v * v; }
  for (int o = 16; o; o >>= 1) sm += __shfl_xor(sm, o, 32);
  __shared__ float red[8];
  if ((tid & 31) == 0) red[tid >> 5] = sm;
  __syncthreads();
  if (tid == 0) {
    float v = 0.f;
#pragma unroll
    for (int w = 0; w < 8; ++w) v += red[w];
    nh[row] = sqrtf(v);
  }
}

// ---------------- weight row-sums ----------------
__global__ void wsum_kernel(const float* __restrict__ dot, const float* __restrict__ nx,
                            const float* __restrict__ nh, float* __restrict__ wsum) {
  int idx = threadIdx.x;  // b*32+s
  int b = idx >> 5;
  float w = 0.f;
#pragma unroll
  for (int t = 0; t < 31; ++t)
    w += dot[(size_t)idx * 31 + t] / (nx[idx] * nh[b * 31 + t]);
  wsum[idx] = w;
}

// ---------------- adaptive pool 2x2 + 2048x3 FC heads ----------------
__global__ __launch_bounds__(256) void head_kernel(
    const float* __restrict__ x_base, const float* __restrict__ scale,
    const float* __restrict__ w, const float* __restrict__ bias,
    float* __restrict__ out0, float* __restrict__ out1) {
  int row = blockIdx.x;
  int tid = threadIdx.x;
  __shared__ float feat[2048];
  const float* x = x_base + (size_t)row * DFLAT;
  float sc = scale ? scale[row] : 1.0f;
  for (int c = tid; c < CHAN; c += 256) {
    const float* xc = x + c * HW;
    float p00 = 0, p01 = 0, p10 = 0, p11 = 0;
#pragma unroll
    for (int ih = 0; ih < 7; ++ih) {
      float r0 = 0, r1 = 0;
#pragma unroll
      for (int iw = 0; iw < 7; ++iw) {
        float v = xc[ih * 7 + iw];
        if (iw < 4) r0 += v;
        if (iw >= 3) r1 += v;
      }
      r0 *= 0.25f; r1 *= 0.25f;
      if (ih < 4) { p00 += r0; p01 += r1; }
      if (ih >= 3) { p10 += r0; p11 += r1; }
    }
    feat[c * 4 + 0] = sc * p00 * 0.25f;
    feat[c * 4 + 1] = sc * p01 * 0.25f;
    feat[c * 4 + 2] = sc * p10 * 0.25f;
    feat[c * 4 + 3] = sc * p11 * 0.25f;
  }
  __syncthreads();
  float a[6] = {0, 0, 0, 0, 0, 0};
  for (int d = tid; d < 2048; d += 256) {
    float fv = feat[d];
#pragma unroll
    for (int ij = 0; ij < 6; ++ij) a[ij] += fv * w[(size_t)ij * 2048 + d];
  }
#pragma unroll
  for (int ij = 0; ij < 6; ++ij)
    for (int o = 16; o; o >>= 1) a[ij] += __shfl_xor(a[ij], o, 32);
  __shared__ float red[8][6];
  if ((tid & 31) == 0) {
#pragma unroll
    for (int ij = 0; ij < 6; ++ij) red[tid >> 5][ij] = a[ij];
  }
  __syncthreads();
  if (tid < 6) {
    float v = 0.f;
#pragma unroll
    for (int wv = 0; wv < 8; ++wv) v += red[wv][tid];
    v += bias[tid];
    int i = tid / 3, j = tid % 3;
    (i == 0 ? out0 : out1)[row * 3 + j] = v;
  }
}

extern "C" void kernel_launch(void* const* d_in, const int* in_sizes, int n_in,
                              void* d_out, int out_size, void* d_ws, size_t ws_size,
                              hipStream_t stream) {
  (void)in_sizes; (void)n_in; (void)out_size; (void)ws_size;
  const float* features = (const float*)d_in[0];
  const float* lstm_w   = (const float*)d_in[1];
  const float* lstm_b   = (const float*)d_in[2];
  const float* vo_w     = (const float*)d_in[3];
  const float* vo_b     = (const float*)d_in[4];
  const float* glob_w   = (const float*)d_in[5];
  const float* glob_b   = (const float*)d_in[6];
  float* out = (float*)d_out;
  char*  ws  = (char*)d_ws;

  bf16_t* A     = (bf16_t*)(ws + OFF_A);
  bf16_t* Bm    = (bf16_t*)(ws + OFF_B);
  float*  gates = (float*)(ws + OFF_G);
  bf16_t* hbf   = (bf16_t*)(ws + OFF_HBF);
  float*  cst   = (float*)(ws + OFF_C);
  float*  hs    = (float*)(ws + OFF_HS);
  float*  dotb  = (float*)(ws + OFF_DOT);
  float*  nx    = (float*)(ws + OFF_NX);
  float*  nh    = (float*)(ws + OFF_NH);
  float*  wsv   = (float*)(ws + OFF_WSUM);

  // one-time bf16 weight image (stays hot in L2 across the 31 steps)
  cvt_bf16_kernel<<<(MDIM * (size_t)KDIM) / 4 / 256, 256, 0, stream>>>(
      lstm_w, A, (size_t)MDIM * KDIM);
  init_state_kernel<<<(CHAN * NREAL + 255) / 256, 256, 0, stream>>>(cst, hbf);

  for (int t = 0; t < 31; ++t) {
    im2col_kernel<<<((size_t)NREAL * KDIM + 255) / 256, 256, 0, stream>>>(
        features, hbf, Bm, t);
    gemm_gates_kernel<<<dim3(MDIM / MT, NPAD / NT), 256, 0, stream>>>(A, Bm, gates);
    lstm_pointwise_kernel<<<(CHAN * NREAL + 255) / 256, 256, 0, stream>>>(
        gates, lstm_b, cst, hbf, hs, t);
  }

  simdot_kernel<<<dim3(SEQ, BATCH), 256, 0, stream>>>(features, hs, dotb, nx);
  nh_kernel<<<BATCH * 31, 256, 0, stream>>>(hs, nh);
  wsum_kernel<<<1, 256, 0, stream>>>(dotb, nx, nh, wsv);

  head_kernel<<<BATCH * 31, 256, 0, stream>>>(hs, nullptr, vo_w, vo_b,
                                              out, out + 744);
  head_kernel<<<BATCH * SEQ, 256, 0, stream>>>(features, wsv, glob_w, glob_b,
                                               out + 1488, out + 2256);
}
